// specific_encoder_8753143349493
// MI455X (gfx1250) — matrix operand order repaired and checked by reading, not compiled, not run on hardware
//
#include <hip/hip_runtime.h>

#define N_NODES 8192
#define DIM_IN  512
#define D_HID   256
#define DOUT    128
#define NEG_SLOPE 0.25f

typedef __attribute__((ext_vector_type(16))) __bf16         v16bf;
typedef __attribute__((ext_vector_type(8)))  unsigned short v8u;
typedef __attribute__((ext_vector_type(8)))  float          v8f;

union FragU { v16bf bf; v8u h[2]; };

__device__ __forceinline__ unsigned short f2bf(float f) {
  unsigned int u = __float_as_uint(f);
  u += 0x7FFFu + ((u >> 16) & 1u);   // round-to-nearest-even
  return (unsigned short)(u >> 16);
}

// pack two floats as bf16 pair: lo in [15:0], hi in [31:16]
__device__ __forceinline__ unsigned int pack2bf(float lo, float hi) {
  return (unsigned int)f2bf(lo) | ((unsigned int)f2bf(hi) << 16);
}

__device__ __forceinline__ float leaky(float v) {
  return v > 0.0f ? v : NEG_SLOPE * v;
}

// ---------------------------------------------------------------------------
// Generic C = A(f32,MxK) @ B(f32,KxN) via bf16 WMMA, optional bias + leaky.
// Block: 256 threads = 8 waves; tile BM=64 x BN=128, BK=32.
// Requires M%64==0, N%128==0, K%32==0 (true for all 5 calls).
// LDS tiles held as u32 (packed bf16 pairs along K); row stride 20 u32 = 80 B
// keeps 16-B alignment for ds_load_b128 fragment reads and staggers banks.
// ---------------------------------------------------------------------------
#define BM 64
#define BN 128
#define BK 32
#define PADW 4   // pad in u32 units (= 8 bf16)
#define ROWW (BK / 2 + PADW)

__global__ __launch_bounds__(256) void gemm_bf16_wmma(
    const float* __restrict__ A, const float* __restrict__ B,
    float* __restrict__ C, const float* __restrict__ bias,
    int M, int N, int K, int fuseLeaky)
{
  __shared__ __attribute__((aligned(16))) unsigned int As32[BM][ROWW];
  __shared__ __attribute__((aligned(16))) unsigned int Bs32[BN][ROWW];

  const int tid  = threadIdx.x;
  const int lane = tid & 31;
  const int wave = tid >> 5;
  const int half = lane >> 4;    // lane 0-15 vs 16-31
  const int l16  = lane & 15;
  const int bm = blockIdx.y * BM;
  const int bn = blockIdx.x * BN;
  const int wm = (wave & 1) * 32;   // wave's M offset in tile
  const int wn = (wave >> 1) * 32;  // wave's N offset in tile

  v8f acc[2][2];
#pragma unroll
  for (int mt = 0; mt < 2; ++mt)
#pragma unroll
    for (int nt = 0; nt < 2; ++nt)
      acc[mt][nt] = v8f{0.f, 0.f, 0.f, 0.f, 0.f, 0.f, 0.f, 0.f};

  for (int k0 = 0; k0 < K; k0 += BK) {
    // --- stage A tile (BM x BK) as packed bf16: 512 float4s / 256 threads ---
#pragma unroll
    for (int i = 0; i < 2; ++i) {
      int idx = tid + 256 * i;
      int r   = idx >> 3;            // 0..63
      int c4  = (idx & 7) << 2;      // k: 0,4,..,28
      const float* ap = A + (size_t)(bm + r) * K + k0 + c4;
      const float4 v = *(const float4*)ap;
      if (k0 + BK < K) __builtin_prefetch(ap + BK, 0, 1);
      As32[r][(c4 >> 1) + 0] = pack2bf(v.x, v.y);
      As32[r][(c4 >> 1) + 1] = pack2bf(v.z, v.w);
    }
    // --- stage B tile (BK x BN) transposed into Bs[n][k], k-pairs packed ---
#pragma unroll
    for (int i = 0; i < 2; ++i) {
      int idx = tid + 256 * i;
      int kp  = idx >> 5;            // k-pair: 0..15 -> k = 2*kp
      int c4  = (idx & 31) << 2;     // n: 0,4,..,124
      const float* bp = B + (size_t)(k0 + 2 * kp) * N + bn + c4;
      const float4 v0 = *(const float4*)bp;        // row k
      const float4 v1 = *(const float4*)(bp + N);  // row k+1
      Bs32[c4 + 0][kp] = pack2bf(v0.x, v1.x);
      Bs32[c4 + 1][kp] = pack2bf(v0.y, v1.y);
      Bs32[c4 + 2][kp] = pack2bf(v0.z, v1.z);
      Bs32[c4 + 3][kp] = pack2bf(v0.w, v1.w);
    }
    __syncthreads();

    // --- load fragments per ISA 7.12.2 layouts ---
    FragU a[2], b[2];
#pragma unroll
    for (int mt = 0; mt < 2; ++mt) {
      const unsigned short* rp =
          (const unsigned short*)&As32[wm + mt * 16 + l16][0];
      // elems 0..7  : K = half*8 + 0..7
      // elems 8..15 : K = 16 + half*8 + 0..7
      a[mt].h[0] = *(const v8u*)(rp + half * 8);
      a[mt].h[1] = *(const v8u*)(rp + 16 + half * 8);
    }
#pragma unroll
    for (int nt = 0; nt < 2; ++nt) {
      const unsigned short* rp =
          (const unsigned short*)&Bs32[wn + nt * 16 + l16][0];
      // elems 0..7  : K = half*8 + 0..7
      // elems 8..15 : K = 16 + half*8 + 0..7
      b[nt].h[0] = *(const v8u*)(rp + half * 8);
      b[nt].h[1] = *(const v8u*)(rp + 16 + half * 8);
    }
#pragma unroll
    for (int mt = 0; mt < 2; ++mt)
#pragma unroll
      for (int nt = 0; nt < 2; ++nt)
        acc[mt][nt] = __builtin_amdgcn_wmma_f32_16x16x32_bf16(
            false, a[mt].bf, false, b[nt].bf, (short)0, acc[mt][nt],
            false, false);
    __syncthreads();
  }

  // --- epilogue: C layout per ISA (VGPR r -> rows half*8+r, col = lane%16) ---
#pragma unroll
  for (int mt = 0; mt < 2; ++mt)
#pragma unroll
    for (int nt = 0; nt < 2; ++nt) {
      const int col = bn + wn + nt * 16 + l16;
      const float bv = bias ? bias[col] : 0.0f;
#pragma unroll
      for (int r = 0; r < 8; ++r) {
        const int row = bm + wm + mt * 16 + half * 8 + r;
        float v = acc[mt][nt][r] + bv;
        if (fuseLeaky) v = leaky(v);
        C[(size_t)row * N + col] = v;
      }
    }
}

// ---------------------------------------------------------------------------
// s1[i] = h[i,:]·a[0:128], s2[i] = h[i,:]·a[128:256]; one wave32 per row.
// ---------------------------------------------------------------------------
__global__ __launch_bounds__(256) void scores_kernel(
    const float* __restrict__ h, const float* __restrict__ a,
    float* __restrict__ s1, float* __restrict__ s2)
{
  const int lane = threadIdx.x & 31;
  const int wave = threadIdx.x >> 5;
  const int row  = blockIdx.x * 8 + wave;
  const float* hr = h + (size_t)row * DOUT;
  float d1 = 0.0f, d2 = 0.0f;
#pragma unroll
  for (int i = 0; i < 4; ++i) {
    const int c = lane + 32 * i;
    const float v = hr[c];
    d1 += v * a[c];
    d2 += v * a[DOUT + c];
  }
#pragma unroll
  for (int off = 16; off > 0; off >>= 1) {
    d1 += __shfl_xor(d1, off, 32);
    d2 += __shfl_xor(d2, off, 32);
  }
  if (lane == 0) { s1[row] = d1; s2[row] = d2; }
}

// ---------------------------------------------------------------------------
// Per-row masked softmax attention + aggregation + final leaky + split out.
// e_ij = leaky(s1[i]+s2[j]); only edges (adj>0) survive the mask in fp32
// (exp(-1e12 - m) == 0), so we softmax over the compacted edge list only.
// Deterministic compaction: per-thread 32-bit edge mask + block prefix sum.
// ---------------------------------------------------------------------------
#define CAP 2048

__global__ __launch_bounds__(256) void attn_kernel(
    const float* __restrict__ adj, const float* __restrict__ h,
    const float* __restrict__ s1, const float* __restrict__ s2,
    float* __restrict__ out)
{
  __shared__ int   ejs[CAP];
  __shared__ float ees[CAP];
  __shared__ int   tscan[256];
  __shared__ float red[8];
  __shared__ float part[DOUT];

  const int i    = blockIdx.x;
  const int tid  = threadIdx.x;
  const int lane = tid & 31;
  const int wave = tid >> 5;
  const float s1i = s1[i];
  const float* __restrict__ arow = adj + (size_t)i * N_NODES;

  // pass 1: bitmask of edges among this thread's 32 strided columns
  unsigned int mask = 0u;
#pragma unroll 8
  for (int it = 0; it < 32; ++it)
    if (arow[tid + (it << 8)] > 0.0f) mask |= (1u << it);

  const int myc = __popc(mask);
  tscan[tid] = myc;
  __syncthreads();
  for (int off = 1; off < 256; off <<= 1) {   // Hillis-Steele inclusive scan
    const int v = (tid >= off) ? tscan[tid - off] : 0;
    __syncthreads();
    tscan[tid] += v;
    __syncthreads();
  }
  const int count = tscan[255];
  const int myoff = tscan[tid] - myc;

  if (count == 0) {
    // all entries masked -> softmax uniform -> h_prime = mean_j h[j,:]
    if (tid < DOUT) {
      float s = 0.0f;
      for (int r = 0; r < N_NODES; ++r) s += h[(size_t)r * DOUT + tid];
      const float v = leaky(s * (1.0f / N_NODES));
      if (tid < 64) out[(size_t)i * 64 + tid] = v;
      else out[(size_t)N_NODES * 64 + (size_t)i * 64 + (tid - 64)] = v;
    }
    return;
  }

  // pass 2: fill compacted edge list; track row max of e
  float lmax = -3.0e38f;
  {
    int p = myoff;
    unsigned int m2 = mask;
    while (m2) {
      const int it = __ffs(m2) - 1;
      m2 &= m2 - 1u;
      const int j = tid + (it << 8);
      const float e = leaky(s1i + s2[j]);
      if (p < CAP) { ejs[p] = j; ees[p] = e; }
      lmax = fmaxf(lmax, e);
      ++p;
    }
  }
#pragma unroll
  for (int off = 16; off > 0; off >>= 1)
    lmax = fmaxf(lmax, __shfl_xor(lmax, off, 32));
  if (lane == 0) red[wave] = lmax;
  __syncthreads();
  float m = red[0];
#pragma unroll
  for (int w = 1; w < 8; ++w) m = fmaxf(m, red[w]);

  const int cnt = count < CAP ? count : CAP;

  // softmax denominator
  float lsum = 0.0f;
  for (int p = tid; p < cnt; p += 256) lsum += expf(ees[p] - m);
#pragma unroll
  for (int off = 16; off > 0; off >>= 1) lsum += __shfl_xor(lsum, off, 32);
  __syncthreads();                    // done reading red[] for m
  if (lane == 0) red[wave] = lsum;
  __syncthreads();
  float S = 0.0f;
#pragma unroll
  for (int w = 0; w < 8; ++w) S += red[w];
  const float invS = 1.0f / S;

  // normalize weights in place
  for (int p = tid; p < cnt; p += 256) ees[p] = expf(ees[p] - m) * invS;
  __syncthreads();

  // aggregate h_prime[c] = sum_e w_e * h[j_e, c]; 2 groups x 128 channels
  const int g = tid >> 7;
  const int c = tid & 127;
  float accv = 0.0f;
  for (int p = g; p < cnt; p += 2)
    accv += ees[p] * h[(size_t)ejs[p] * DOUT + c];
  if (g == 1) part[c] = accv;
  __syncthreads();
  if (g == 0) {
    const float v = leaky(accv + part[c]);
    if (c < 64) out[(size_t)i * 64 + c] = v;
    else out[(size_t)N_NODES * 64 + (size_t)i * 64 + (c - 64)] = v;
  }
}

// ---------------------------------------------------------------------------
extern "C" void kernel_launch(void* const* d_in, const int* in_sizes, int n_in,
                              void* d_out, int out_size, void* d_ws, size_t ws_size,
                              hipStream_t stream)
{
  (void)in_sizes; (void)n_in; (void)out_size; (void)ws_size;
  const float* x   = (const float*)d_in[0];
  const float* adj = (const float*)d_in[1];
  const float* W1  = (const float*)d_in[2];
  const float* b1  = (const float*)d_in[3];
  const float* W2  = (const float*)d_in[4];
  const float* b2  = (const float*)d_in[5];
  const float* Wg  = (const float*)d_in[6];
  const float* a   = (const float*)d_in[7];
  float* out = (float*)d_out;

  float* t1 = (float*)d_ws;                         // 8192x256
  float* x1 = t1 + (size_t)N_NODES * D_HID;         // 8192x256
  float* t2 = x1 + (size_t)N_NODES * D_HID;         // 8192x256
  float* x2 = t2 + (size_t)N_NODES * D_HID;         // 8192x256
  float* h  = x2 + (size_t)N_NODES * D_HID;         // 8192x128
  float* s1 = h  + (size_t)N_NODES * DOUT;          // 8192
  float* s2 = s1 + N_NODES;                         // 8192

  const dim3 blk(256);
  // t1 = x @ W1
  gemm_bf16_wmma<<<dim3(D_HID / BN, N_NODES / BM), blk, 0, stream>>>(
      x, W1, t1, nullptr, N_NODES, D_HID, DIM_IN, 0);
  // x1 = leaky(adj @ t1 + b1)
  gemm_bf16_wmma<<<dim3(D_HID / BN, N_NODES / BM), blk, 0, stream>>>(
      adj, t1, x1, b1, N_NODES, D_HID, N_NODES, 1);
  // t2 = x1 @ W2
  gemm_bf16_wmma<<<dim3(D_HID / BN, N_NODES / BM), blk, 0, stream>>>(
      x1, W2, t2, nullptr, N_NODES, D_HID, D_HID, 0);
  // x2 = leaky(adj @ t2 + b2)
  gemm_bf16_wmma<<<dim3(D_HID / BN, N_NODES / BM), blk, 0, stream>>>(
      adj, t2, x2, b2, N_NODES, D_HID, N_NODES, 1);
  // h = x2 @ Wg
  gemm_bf16_wmma<<<dim3(DOUT / BN, N_NODES / BM), blk, 0, stream>>>(
      x2, Wg, h, nullptr, N_NODES, DOUT, D_HID, 0);
  // s1 = h @ a1, s2 = h @ a2
  scores_kernel<<<dim3(N_NODES / 8), blk, 0, stream>>>(h, a, s1, s2);
  // masked softmax attention + aggregation + output split
  attn_kernel<<<dim3(N_NODES), blk, 0, stream>>>(adj, h, s1, s2, out);
}
